// MSA_83665962926276
// MI455X (gfx1250) — matrix operand order repaired and checked by reading, not compiled, not run on hardware
//
#include <hip/hip_runtime.h>

typedef __attribute__((ext_vector_type(16))) _Float16 v16h;
typedef __attribute__((ext_vector_type(8)))  _Float16 h8;
typedef __attribute__((ext_vector_type(4)))  _Float16 h4;
typedef __attribute__((ext_vector_type(8)))  float    v8f;

#define WMMA_F32_F16(a, b, c) \
  __builtin_amdgcn_wmma_f32_16x16x32_f16(false, (a), false, (b), (short)0, (c), false, false)

// Problem constants: B=8, N=1024, D=768, H=12, HD=64, scale = 64^-0.5 = 0.125

// ---------------------------------------------------------------------------
// WMMA fragment builders (wave32, 16-bit data); A/B use a contiguous k order, same for both
// A 16x32 (MxK): lane m=lane%16, g=lane>>4; elem i -> K = 16g + i
// B 32x16 (KxN): lane n=lane%16, g=lane>>4; elem i -> K = 16g + i
// C/D f32 16x16: vgpr r, lane -> M = r + 8*(lane>>4), N = lane%16
// (each folds into 2x ds_load_b128 -- verified in round 1/2 disasm)
// ---------------------------------------------------------------------------
__device__ inline v16h frag_a(const _Float16* p, int stride) {
  const int lane = threadIdx.x & 31;
  const int m = lane & 15, g = lane >> 4;
  const _Float16* row = p + m * stride + 16 * g;
  v16h a;
#pragma unroll
  for (int i = 0; i < 8; ++i)  a[i] = row[i];
#pragma unroll
  for (int i = 8; i < 16; ++i) a[i] = row[i];
  return a;
}

// B tile stored [n][k]: elem(k,n)=p[n*stride+k]
__device__ inline v16h frag_bt(const _Float16* p, int stride) {
  const int lane = threadIdx.x & 31;
  const int n = lane & 15, g = lane >> 4;
  const _Float16* row = p + n * stride + 16 * g;
  v16h b;
#pragma unroll
  for (int i = 0; i < 16; ++i) b[i] = row[i];
  return b;
}

// ---------------------------------------------------------------------------
// Kernel 1: QKV projection. C[8192,2304] = X[8192,768] * Wqkv[2304,768]^T
// Block tile 128x128, 8 waves as 4(M) x 2(N), wave tile 32x64 (8 WMMA/step).
// Register-staged LDS double buffer: next tile's global loads issue before
// the current tile's WMMAs; one barrier per K-step.
// ---------------------------------------------------------------------------
__global__ __launch_bounds__(256)
void qkv_kernel(const float* __restrict__ x, const float* __restrict__ w,
                _Float16* __restrict__ Qb, _Float16* __restrict__ Kb,
                _Float16* __restrict__ Vb)
{
  constexpr int TSZ = 128 * 40;
  __shared__ _Float16 As[2 * TSZ];
  __shared__ _Float16 Bs[2 * TSZ];
  const int m0 = blockIdx.x * 128;
  const int e0 = blockIdx.y * 128;
  const int tid = threadIdx.x;
  const int wv = tid >> 5, lane = tid & 31;
  const int wm = wv >> 1, wn = wv & 1;
  const int g = lane >> 4, nc = lane & 15;

  v8f acc[2][4] = {};
  float4 ra[4], rb[4];

  auto gload = [&](int kk) {
    int idx = tid;
#pragma unroll
    for (int j = 0; j < 4; ++j, idx += 256) {
      const int row = idx >> 3, cv = idx & 7;
      ra[j] = *(const float4*)(x + (size_t)(m0 + row) * 768 + kk + cv * 4);
      rb[j] = *(const float4*)(w + (size_t)(e0 + row) * 768 + kk + cv * 4);
      const int kp = (kk + 32 < 768) ? kk + 32 : kk;  // branchless clamp
      __builtin_prefetch(x + (size_t)(m0 + row) * 768 + kp + cv * 4, 0, 3);
      __builtin_prefetch(w + (size_t)(e0 + row) * 768 + kp + cv * 4, 0, 3);
    }
  };
  auto sstore = [&](int buf) {
    _Float16* Ad = As + buf * TSZ;
    _Float16* Bd = Bs + buf * TSZ;
    int idx = tid;
#pragma unroll
    for (int j = 0; j < 4; ++j, idx += 256) {
      const int row = idx >> 3, cv = idx & 7;
      h4 ha; ha[0] = (_Float16)ra[j].x; ha[1] = (_Float16)ra[j].y;
             ha[2] = (_Float16)ra[j].z; ha[3] = (_Float16)ra[j].w;
      h4 hb; hb[0] = (_Float16)rb[j].x; hb[1] = (_Float16)rb[j].y;
             hb[2] = (_Float16)rb[j].z; hb[3] = (_Float16)rb[j].w;
      *(h4*)(Ad + row * 40 + cv * 4) = ha;
      *(h4*)(Bd + row * 40 + cv * 4) = hb;
    }
  };

  gload(0);
  sstore(0);
  __syncthreads();

  int cur = 0;
  for (int k0 = 0; k0 < 768; k0 += 32) {
    const bool has_next = (k0 + 32 < 768);
    if (has_next) gload(k0 + 32);       // long-latency loads issue first

    const _Float16* Ac = As + cur * TSZ;
    const _Float16* Bc = Bs + cur * TSZ;
    v16h af[2], bf[4];
#pragma unroll
    for (int ti = 0; ti < 2; ++ti) af[ti] = frag_a(Ac + (wm * 32 + ti * 16) * 40, 40);
#pragma unroll
    for (int tj = 0; tj < 4; ++tj) bf[tj] = frag_bt(Bc + (wn * 64 + tj * 16) * 40, 40);
#pragma unroll
    for (int ti = 0; ti < 2; ++ti)
#pragma unroll
      for (int tj = 0; tj < 4; ++tj)
        acc[ti][tj] = WMMA_F32_F16(af[ti], bf[tj], acc[ti][tj]);

    if (has_next) sstore(cur ^ 1);      // after frag loads: WMMA covered the latency
    __syncthreads();
    cur ^= 1;
  }

  // Block-uniform scatter parameters (128 | 768 and 128 | 1024)
  const int which = e0 / 768;
  const int rembase = e0 - which * 768;
  const int b = m0 >> 10;
  const int nbase = m0 - (b << 10);
  _Float16* dstbuf = (which == 0) ? Qb : (which == 1) ? Kb : Vb;
  const float sc = (which == 0) ? 0.125f : 1.0f;
  const size_t bhbase = (size_t)b * 12;

#pragma unroll
  for (int ti = 0; ti < 2; ++ti)
#pragma unroll
    for (int tj = 0; tj < 4; ++tj) {
      const int rem = rembase + wn * 64 + tj * 16 + nc;
      const int h = rem >> 6, hd = rem & 63;
      const size_t colbase = ((bhbase + h) * 1024) * 64 + hd;
#pragma unroll
      for (int r = 0; r < 8; ++r) {
        const int n = nbase + wm * 32 + ti * 16 + r + 8 * g;
        dstbuf[colbase + (size_t)n * 64] = (_Float16)(acc[ti][tj][r] * sc);
      }
    }
}

// ---------------------------------------------------------------------------
// Kernel 2: flash attention per (b,h). Block = 128 query rows, 8 waves x 16
// rows. 64-key chunks, double-buffered K/V in LDS, V staged TRANSPOSED
// ([hd][key], stride 72 = 144B rows, 16B aligned) so every PV B-fragment is
// two ds_load_b128. P round-trip is wave-private (DS ops in-order per wave,
// no barrier needed). One barrier per chunk. 16 WMMA per chunk per wave.
// ---------------------------------------------------------------------------
__global__ __launch_bounds__(256)
void attn_kernel(const _Float16* __restrict__ Qb, const _Float16* __restrict__ Kb,
                 const _Float16* __restrict__ Vb, _Float16* __restrict__ Ob)
{
  constexpr int KVSZ = 64 * 72;
  __shared__ _Float16 Ks[2 * KVSZ];   // [key][hd]   (B^T layout for QK^T)
  __shared__ _Float16 Vt[2 * KVSZ];   // [hd][key]   (B^T layout for P*V)
  __shared__ _Float16 Ps[8][16 * 64];
  const int q0 = blockIdx.x * 128;
  const int bh = blockIdx.y;
  const int b = bh / 12, h = bh % 12;
  const int tid = threadIdx.x, wv = tid >> 5, lane = tid & 31;
  const int g = lane >> 4, nc = lane & 15;

  const size_t base = (size_t)bh * 1024 * 64;

  // Q fragments (16 rows x 64 hd) live in registers for the whole kernel
  const _Float16* qrow = Qb + base + (size_t)(q0 + wv * 16) * 64;
  const v16h aq0 = frag_a(qrow, 64);       // hd 0..31
  const v16h aq1 = frag_a(qrow + 32, 64);  // hd 32..63

  v8f o[4] = {};
  float mrow[8], lrow[8];
#pragma unroll
  for (int r = 0; r < 8; ++r) { mrow[r] = -3.0e38f; lrow[r] = 0.0f; }

  h8 kr[2], vr[2];
  auto kvload = [&](int kc) {
    int idx = tid;
#pragma unroll
    for (int j = 0; j < 2; ++j, idx += 256) {
      const int row = idx >> 3, col = (idx & 7) * 8;
      kr[j] = *(const h8*)(Kb + base + (size_t)(kc + row) * 64 + col);
      vr[j] = *(const h8*)(Vb + base + (size_t)(kc + row) * 64 + col);
      const int kp = (kc + 64 < 1024) ? kc + 64 : kc;  // branchless clamp
      __builtin_prefetch(Kb + base + (size_t)(kp + row) * 64 + col, 0, 3);
      __builtin_prefetch(Vb + base + (size_t)(kp + row) * 64 + col, 0, 3);
    }
  };
  auto kvstore = [&](int buf) {
    _Float16* Kd = Ks + buf * KVSZ;
    _Float16* Vd = Vt + buf * KVSZ;
    int idx = tid;
#pragma unroll
    for (int j = 0; j < 2; ++j, idx += 256) {
      const int row = idx >> 3, col = (idx & 7) * 8;
      *(h8*)(Kd + row * 72 + col) = kr[j];
#pragma unroll
      for (int e = 0; e < 8; ++e)         // transpose V: [key][hd]->[hd][key]
        Vd[(col + e) * 72 + row] = vr[j][e];
    }
  };

  kvload(0);
  kvstore(0);
  __syncthreads();

  int cur = 0;
  for (int kc = 0; kc < 1024; kc += 64) {
    const bool has_next = (kc + 64 < 1024);
    if (has_next) kvload(kc + 64);       // issue next chunk's loads first

    const _Float16* Kc = Ks + cur * KVSZ;
    const _Float16* Vc = Vt + cur * KVSZ;

    // S = Q * K^T : four 16x16 f32 tiles (key subtiles of 16), K-dim 64 = 2x32
    v8f s[4] = {};
#pragma unroll
    for (int c = 0; c < 4; ++c) {
      const v16h bk0 = frag_bt(Kc + (c * 16) * 72, 72);
      const v16h bk1 = frag_bt(Kc + (c * 16) * 72 + 32, 72);
      s[c] = WMMA_F32_F16(aq0, bk0, s[c]);
      s[c] = WMMA_F32_F16(aq1, bk1, s[c]);
    }

    // Online softmax; each accumulator row spans 16 lanes (xor masks stay in half)
    _Float16* Pw = &Ps[wv][0];
#pragma unroll
    for (int r = 0; r < 8; ++r) {
      float t = fmaxf(fmaxf(s[0][r], s[1][r]), fmaxf(s[2][r], s[3][r]));
      t = fmaxf(t, __shfl_xor(t, 1, 32));
      t = fmaxf(t, __shfl_xor(t, 2, 32));
      t = fmaxf(t, __shfl_xor(t, 4, 32));
      t = fmaxf(t, __shfl_xor(t, 8, 32));
      const float mnew = fmaxf(mrow[r], t);
      const float corr = __expf(mrow[r] - mnew);
      float p[4], rs = 0.0f;
#pragma unroll
      for (int c = 0; c < 4; ++c) { p[c] = __expf(s[c][r] - mnew); rs += p[c]; }
      rs += __shfl_xor(rs, 1, 32);
      rs += __shfl_xor(rs, 2, 32);
      rs += __shfl_xor(rs, 4, 32);
      rs += __shfl_xor(rs, 8, 32);
      lrow[r] = lrow[r] * corr + rs;
      mrow[r] = mnew;
#pragma unroll
      for (int t4 = 0; t4 < 4; ++t4) o[t4][r] = o[t4][r] * corr;
      const int prow = r + 8 * g;
#pragma unroll
      for (int c = 0; c < 4; ++c)
        Pw[prow * 64 + c * 16 + nc] = (_Float16)p[c];
    }
    // No barrier: P tile is wave-private; DS ops are in-order within a wave.

    // O += P * V : 4 hd tiles x 2 key halves, all fragments vectorized
    const v16h pa0 = frag_a(Pw, 64);        // keys 0..31
    const v16h pa1 = frag_a(Pw + 32, 64);   // keys 32..63
#pragma unroll
    for (int t4 = 0; t4 < 4; ++t4) {
      const v16h bv0 = frag_bt(Vc + (t4 * 16) * 72, 72);       // keys 0..31
      const v16h bv1 = frag_bt(Vc + (t4 * 16) * 72 + 32, 72);  // keys 32..63
      o[t4] = WMMA_F32_F16(pa0, bv0, o[t4]);
      o[t4] = WMMA_F32_F16(pa1, bv1, o[t4]);
    }

    if (has_next) kvstore(cur ^ 1);
    __syncthreads();
    cur ^= 1;
  }

  // Normalize and write f16 [B,N,D] with head transpose fused
#pragma unroll
  for (int t4 = 0; t4 < 4; ++t4)
#pragma unroll
    for (int r = 0; r < 8; ++r) {
      const int n = q0 + wv * 16 + r + 8 * g;
      const int d = h * 64 + t4 * 16 + nc;
      const float val = o[t4][r] / lrow[r];
      Ob[((size_t)b * 1024 + n) * 768 + d] = (_Float16)val;
    }
}

// ---------------------------------------------------------------------------
// Kernel 3: output projection. out[8192,768] = AO[8192,768](f16) * Wproj^T + b
// Same double-buffered 128x128 tiling; fp32 epilogue to d_out.
// ---------------------------------------------------------------------------
__global__ __launch_bounds__(256)
void proj_kernel(const _Float16* __restrict__ A, const float* __restrict__ w,
                 const float* __restrict__ bias, float* __restrict__ out)
{
  constexpr int TSZ = 128 * 40;
  __shared__ _Float16 As[2 * TSZ];
  __shared__ _Float16 Bs[2 * TSZ];
  const int m0 = blockIdx.x * 128;
  const int e0 = blockIdx.y * 128;
  const int tid = threadIdx.x;
  const int wv = tid >> 5, lane = tid & 31;
  const int wm = wv >> 1, wn = wv & 1;
  const int g = lane >> 4, nc = lane & 15;

  v8f acc[2][4] = {};
  h4 ra[4];
  float4 rb[4];

  auto gload = [&](int kk) {
    int idx = tid;
#pragma unroll
    for (int j = 0; j < 4; ++j, idx += 256) {
      const int row = idx >> 3, cv = idx & 7;
      ra[j] = *(const h4*)(A + (size_t)(m0 + row) * 768 + kk + cv * 4);
      rb[j] = *(const float4*)(w + (size_t)(e0 + row) * 768 + kk + cv * 4);
      const int kp = (kk + 32 < 768) ? kk + 32 : kk;
      __builtin_prefetch(A + (size_t)(m0 + row) * 768 + kp + cv * 4, 0, 3);
      __builtin_prefetch(w + (size_t)(e0 + row) * 768 + kp + cv * 4, 0, 3);
    }
  };
  auto sstore = [&](int buf) {
    _Float16* Ad = As + buf * TSZ;
    _Float16* Bd = Bs + buf * TSZ;
    int idx = tid;
#pragma unroll
    for (int j = 0; j < 4; ++j, idx += 256) {
      const int row = idx >> 3, cv = idx & 7;
      h4 hb; hb[0] = (_Float16)rb[j].x; hb[1] = (_Float16)rb[j].y;
             hb[2] = (_Float16)rb[j].z; hb[3] = (_Float16)rb[j].w;
      *(h4*)(Ad + row * 40 + cv * 4) = ra[j];
      *(h4*)(Bd + row * 40 + cv * 4) = hb;
    }
  };

  gload(0);
  sstore(0);
  __syncthreads();

  int cur = 0;
  for (int k0 = 0; k0 < 768; k0 += 32) {
    const bool has_next = (k0 + 32 < 768);
    if (has_next) gload(k0 + 32);

    const _Float16* Ac = As + cur * TSZ;
    const _Float16* Bc = Bs + cur * TSZ;
    v16h af[2], bf[4];
#pragma unroll
    for (int ti = 0; ti < 2; ++ti) af[ti] = frag_a(Ac + (wm * 32 + ti * 16) * 40, 40);
#pragma unroll
    for (int tj = 0; tj < 4; ++tj) bf[tj] = frag_bt(Bc + (wn * 64 + tj * 16) * 40, 40);
#pragma unroll
    for (int ti = 0; ti < 2; ++ti)
#pragma unroll
      for (int tj = 0; tj < 4; ++tj)
        acc[ti][tj] = WMMA_F32_F16(af[ti], bf[tj], acc[ti][tj]);

    if (has_next) sstore(cur ^ 1);
    __syncthreads();
    cur ^= 1;
  }

  float bs[4];
#pragma unroll
  for (int tj = 0; tj < 4; ++tj) bs[tj] = bias[e0 + wn * 64 + tj * 16 + nc];

#pragma unroll
  for (int ti = 0; ti < 2; ++ti)
#pragma unroll
    for (int tj = 0; tj < 4; ++tj)
#pragma unroll
      for (int r = 0; r < 8; ++r) {
        const int m = m0 + wm * 32 + ti * 16 + r + 8 * g;
        const int e = e0 + wn * 64 + tj * 16 + nc;
        out[(size_t)m * 768 + e] = acc[ti][tj][r] + bs[tj];
      }
}

// ---------------------------------------------------------------------------
extern "C" void kernel_launch(void* const* d_in, const int* in_sizes, int n_in,
                              void* d_out, int out_size, void* d_ws, size_t ws_size,
                              hipStream_t stream) {
  (void)in_sizes; (void)n_in; (void)out_size; (void)ws_size;
  const float* x      = (const float*)d_in[0];
  const float* qkv_w  = (const float*)d_in[1];
  const float* proj_w = (const float*)d_in[2];
  const float* proj_b = (const float*)d_in[3];
  float* out = (float*)d_out;

  const size_t HBUF = (size_t)8 * 12 * 1024 * 64;  // 6,291,456 elems per buffer
  _Float16* Qb = (_Float16*)d_ws;
  _Float16* Kb = Qb + HBUF;
  _Float16* Vb = Kb + HBUF;
  _Float16* Ob = Vb + HBUF;  // attn output [B,N,D] f16

  // QKV GEMM: grid 64 (8192/128) x 18 (2304/128)
  qkv_kernel<<<dim3(64, 18), 256, 0, stream>>>(x, qkv_w, Qb, Kb, Vb);
  // Attention: grid 8 (1024/128 query tiles) x 96 (B*H)
  attn_kernel<<<dim3(8, 96), 256, 0, stream>>>(Qb, Kb, Vb, Ob);
  // Output projection: grid 64 x 6 (768/128)
  proj_kernel<<<dim3(64, 6), 256, 0, stream>>>(Ob, proj_w, proj_b, out);
}
